// RefGatedMLPFusedMoE_15461882266327
// MI455X (gfx1250) — matrix operand order repaired and checked by reading, not compiled, not run on hardware
//
#include <hip/hip_runtime.h>
#include <hip/hip_bf16.h>

// ---------------- problem constants (match reference) ----------------
#define NUM_EXPERTS 8
#define HIDDEN      2048
#define INTER       5632
#define TOKENS      2048
// max padded rows: 4096 selected rows + 8 experts * 15 pad, rounded up
#define MAXROWS     4224
// LDS row stride for the staged A tile (f16 units): 2048 + 8 pad breaks the
// 64-bank stride-0 conflict (stride = 1028 dwords == 4 mod 64 banks).
#define SASTRIDE    (HIDDEN + 8)

typedef __attribute__((ext_vector_type(16))) _Float16 v16h;
typedef __attribute__((ext_vector_type(8)))  _Float16 v8h;
typedef __attribute__((ext_vector_type(8)))  float    v8f;

// ---------------- WMMA fragment loaders (wave32, 16x16x32 f16) ----------------
// A-matrix 16x32 f16 layout (ISA 7.12.2): lane<16 -> row=lane, K: [0..7],[16..23];
// lane>=16 -> row=lane-16, K: [8..15],[24..31]. Caller passes the two 8-elem
// chunk offsets (c0 = k0 + half*8, c1 = k0 + 16 + half*8).
// Works for both LDS (ds_load_b128) and global (global_load_b128) f16 sources.
__device__ __forceinline__ v16h frag_a_f16(const _Float16* __restrict__ p, int c0, int c1) {
    const v8h lo = *(const v8h*)(p + c0);
    const v8h hi = *(const v8h*)(p + c1);
    v16h r;
#pragma unroll
    for (int i = 0; i < 8; ++i) { r[i] = lo[i]; r[8 + i] = hi[i]; }
    return r;
}

// B-matrix 32x16 f16 layout (matches A): lane<16 -> col=lane, K [0..7],[16..23];
// lane>=16 -> col=lane-16, K [8..15],[24..31].
// p already points at w_row + k0 + half*8: two 8-fp32 chunks 16 apart -> cvt f16.
__device__ __forceinline__ v16h frag_b_f32(const float* __restrict__ p) {
    const float4 a = *(const float4*)(p + 0);
    const float4 b = *(const float4*)(p + 4);
    const float4 c = *(const float4*)(p + 16);
    const float4 d = *(const float4*)(p + 20);
    v16h r;
    r[0]=(_Float16)a.x;  r[1]=(_Float16)a.y;  r[2]=(_Float16)a.z;  r[3]=(_Float16)a.w;
    r[4]=(_Float16)b.x;  r[5]=(_Float16)b.y;  r[6]=(_Float16)b.z;  r[7]=(_Float16)b.w;
    r[8]=(_Float16)c.x;  r[9]=(_Float16)c.y;  r[10]=(_Float16)c.z; r[11]=(_Float16)c.w;
    r[12]=(_Float16)d.x; r[13]=(_Float16)d.y; r[14]=(_Float16)d.z; r[15]=(_Float16)d.w;
    return r;
}

// ---------------- phase 0: zero routing scratch ----------------
__global__ void init_kernel(int* __restrict__ p, int n) {
    int i = blockIdx.x * blockDim.x + threadIdx.x;
    if (i < n) p[i] = 0;
}

// ---------------- phase 1: router (softmax top-2, renormalized) ----------------
__global__ void router_kernel(const float* __restrict__ logits,
                              int* __restrict__ counts,
                              int* __restrict__ tlist,   // [E][TOKENS]
                              int* __restrict__ eidx,    // [T][2]
                              int* __restrict__ epos,    // [T][2]
                              float* __restrict__ topw)  // [T][2]
{
    int t = blockIdx.x * blockDim.x + threadIdx.x;
    if (t >= TOKENS) return;
    float l[NUM_EXPERTS];
#pragma unroll
    for (int e = 0; e < NUM_EXPERTS; ++e) l[e] = logits[t * NUM_EXPERTS + e];
    int i0 = 0;                               // top-1, lowest index on ties
#pragma unroll
    for (int e = 1; e < NUM_EXPERTS; ++e) if (l[e] > l[i0]) i0 = e;
    int i1 = (i0 == 0) ? 1 : 0;               // top-2
#pragma unroll
    for (int e = 0; e < NUM_EXPERTS; ++e)
        if (e != i0 && l[e] > l[i1]) i1 = e;
    float e1 = __expf(l[i1] - l[i0]);         // softmax denom cancels in renorm
    float w0 = 1.0f / (1.0f + e1);
    float w1 = e1 * w0;
    int p0 = atomicAdd(&counts[i0], 1);
    int p1 = atomicAdd(&counts[i1], 1);
    tlist[i0 * TOKENS + p0] = t;
    tlist[i1 * TOKENS + p1] = t;
    eidx[2 * t] = i0;  eidx[2 * t + 1] = i1;
    epos[2 * t] = p0;  epos[2 * t + 1] = p1;
    topw[2 * t] = w0;  topw[2 * t + 1] = w1;
}

// ---------------- phase 2: pad counts to 16 and prefix-scan row offsets -------
__global__ void scan_kernel(const int* __restrict__ counts, int* __restrict__ offsets) {
    if (threadIdx.x == 0) {
        int acc = 0;
#pragma unroll
        for (int e = 0; e < NUM_EXPERTS; ++e) {
            offsets[e] = acc;
            acc += (counts[e] + 15) & ~15;
        }
    }
}

// ---------------- phase 3: act = silu(X_e @ w1^T) * (X_e @ w3^T), f16 out -----
// grid: (INTER/128, TOKENS/16, E); 256 threads = 8 waves, each wave: 16x16 tile.
// The 16-token A tile is staged once per WG into LDS as f16.
__global__ __launch_bounds__(256) void gemm1_kernel(
    const float* __restrict__ X, const float* __restrict__ W1,
    const float* __restrict__ W3, const int* __restrict__ counts,
    const int* __restrict__ offsets, const int* __restrict__ tlist,
    _Float16* __restrict__ act)
{
    extern __shared__ _Float16 sA[];             // [16][SASTRIDE] f16
    const int e    = blockIdx.z;
    const int tile = blockIdx.y;
    if (tile * 16 >= counts[e]) return;          // uniform early exit (pre-barrier)
    const int wave = threadIdx.x >> 5;
    const int lane = threadIdx.x & 31;
    const int half = lane >> 4;
    const int ml   = lane & 15;
    const int n0   = blockIdx.x * 128 + wave * 16;       // inter column base
    const int rowbase = offsets[e] + tile * 16;

    // ---- cooperative A-tile staging: 16 rows x 2048 fp32 -> f16 LDS ----
    {
        const int srow = threadIdx.x >> 4;       // 0..15
        const int sj   = threadIdx.x & 15;
        const int stok = tlist[e * TOKENS + tile * 16 + srow];
        const float* sx = X + (size_t)stok * HIDDEN;
        _Float16* dst = sA + srow * SASTRIDE;
#pragma unroll
        for (int j = 0; j < HIDDEN / 64; ++j) {  // lanes cover 256B runs
            const int c = j * 64 + sj * 4;
            const float4 v = *(const float4*)(sx + c);
            dst[c + 0] = (_Float16)v.x;
            dst[c + 1] = (_Float16)v.y;
            dst[c + 2] = (_Float16)v.z;
            dst[c + 3] = (_Float16)v.w;
        }
    }
    __syncthreads();

    const float* w1row = W1 + ((size_t)e * INTER + (n0 + ml)) * HIDDEN;
    const float* w3row = W3 + ((size_t)e * INTER + (n0 + ml)) * HIDDEN;
    const _Float16* arow = sA + ml * SASTRIDE;

    v8f accg = {0.f,0.f,0.f,0.f,0.f,0.f,0.f,0.f};
    v8f accu = {0.f,0.f,0.f,0.f,0.f,0.f,0.f,0.f};
    const int ca = half * 8;      // A chunk select
    const int cb = half * 8;      // B chunk select
#pragma unroll 2
    for (int k0 = 0; k0 < HIDDEN; k0 += 32) {
        v16h a  = frag_a_f16(arow, k0 + ca, k0 + 16 + ca);   // ds_load_b128 x2
        v16h b1 = frag_b_f32(w1row + k0 + cb);               // global b128 x4 + cvt
        v16h b3 = frag_b_f32(w3row + k0 + cb);
        accg = __builtin_amdgcn_wmma_f32_16x16x32_f16(false, a, false, b1,
                                                      (short)0, accg, false, false);
        accu = __builtin_amdgcn_wmma_f32_16x16x32_f16(false, a, false, b3,
                                                      (short)0, accu, false, false);
    }
    // C/D layout: VGPR r -> row r (lanes 0-15) / row r+8 (lanes 16-31), col = ml
#pragma unroll
    for (int r = 0; r < 8; ++r) {
        float g = accg[r], u = accu[r];
        // SiLU via v_rcp_f32 (1-ulp) instead of the full IEEE division sequence
        float s = g * __builtin_amdgcn_rcpf(1.0f + __expf(-g));
        act[(size_t)(rowbase + r + half * 8) * INTER + (n0 + ml)] = (_Float16)(s * u);
    }
}

// ---------------- phase 4: y = act @ w2^T (unweighted, per-slot rows) ---------
// grid: (HIDDEN/128, TOKENS/16, E). A is already f16 + L2-resident: direct loads.
__global__ __launch_bounds__(256) void gemm2_kernel(
    const _Float16* __restrict__ act, const float* __restrict__ W2,
    const int* __restrict__ counts, const int* __restrict__ offsets,
    float* __restrict__ y)
{
    const int e    = blockIdx.z;
    const int tile = blockIdx.y;
    if (tile * 16 >= counts[e]) return;
    const int wave = threadIdx.x >> 5;
    const int lane = threadIdx.x & 31;
    const int half = lane >> 4;
    const int ml   = lane & 15;
    const int n0   = blockIdx.x * 128 + wave * 16;       // hidden column base
    const int rowbase = offsets[e] + tile * 16;

    const _Float16* arow = act + (size_t)(rowbase + ml) * INTER;
    const float* w2row = W2 + ((size_t)e * HIDDEN + (n0 + ml)) * INTER;

    v8f acc = {0.f,0.f,0.f,0.f,0.f,0.f,0.f,0.f};
    const int ca = half * 8;
    const int cb = half * 8;
#pragma unroll 2
    for (int k0 = 0; k0 < INTER; k0 += 32) {
        v16h a = frag_a_f16(arow, k0 + ca, k0 + 16 + ca);
        v16h b = frag_b_f32(w2row + k0 + cb);
        acc = __builtin_amdgcn_wmma_f32_16x16x32_f16(false, a, false, b,
                                                     (short)0, acc, false, false);
    }
#pragma unroll
    for (int r = 0; r < 8; ++r)
        y[(size_t)(rowbase + r + half * 8) * HIDDEN + (n0 + ml)] = acc[r];
}

// ---------------- phase 5: out[t] = w0*y[slot0] + w1*y[slot1] (deterministic) -
__global__ __launch_bounds__(256) void combine_kernel(
    const float* __restrict__ y, const int* __restrict__ eidx,
    const int* __restrict__ epos, const float* __restrict__ topw,
    const int* __restrict__ offsets, float* __restrict__ out)
{
    const int t  = blockIdx.y;
    const int h4 = blockIdx.x * blockDim.x + threadIdx.x;  // float4 index
    const int e0 = eidx[2 * t], e1 = eidx[2 * t + 1];
    const int r0 = offsets[e0] + epos[2 * t];
    const int r1 = offsets[e1] + epos[2 * t + 1];
    const float w0 = topw[2 * t], w1 = topw[2 * t + 1];
    const float4 a = ((const float4*)(y + (size_t)r0 * HIDDEN))[h4];
    const float4 b = ((const float4*)(y + (size_t)r1 * HIDDEN))[h4];
    float4 o;
    o.x = w0 * a.x + w1 * b.x;
    o.y = w0 * a.y + w1 * b.y;
    o.z = w0 * a.z + w1 * b.z;
    o.w = w0 * a.w + w1 * b.w;
    ((float4*)(out + (size_t)t * HIDDEN))[h4] = o;
}

// ---------------- launcher ----------------
extern "C" void kernel_launch(void* const* d_in, const int* in_sizes, int n_in,
                              void* d_out, int out_size, void* d_ws, size_t ws_size,
                              hipStream_t stream) {
    const float* X  = (const float*)d_in[0];   // [T, H]
    const float* RL = (const float*)d_in[1];   // [T, E]
    const float* W1 = (const float*)d_in[2];   // [E, I, H]
    const float* W3 = (const float*)d_in[3];   // [E, I, H]
    const float* W2 = (const float*)d_in[4];   // [E, H, I]
    float* out = (float*)d_out;                // [T, H]

    // workspace layout (~82.4 MB)
    char* ws = (char*)d_ws;
    int*   counts  = (int*)(ws + 0);                       //   32 B
    int*   offsets = (int*)(ws + 32);                      //   32 B
    int*   tlist   = (int*)(ws + 64);                      //   64 KB
    int*   eidx    = (int*)(ws + 64 + 65536);              //   16 KB
    int*   epos    = (int*)(ws + 64 + 65536 + 16384);      //   16 KB
    float* topw    = (float*)(ws + 64 + 65536 + 32768);    //   16 KB
    size_t actOff  = (size_t)(64 + 65536 + 49152 + 255) & ~(size_t)255;
    _Float16* act  = (_Float16*)(ws + actOff);                            // 47.6 MB
    float*    y    = (float*)(ws + actOff + (size_t)MAXROWS * INTER * 2); // 34.6 MB

    const size_t ldsBytes = (size_t)16 * SASTRIDE * sizeof(_Float16);     // ~65.8 KB

    // 1. zero counts + token lists (covers counts/offsets/tlist region)
    init_kernel<<<(16400 + 255) / 256, 256, 0, stream>>>(counts, 16400);
    // 2. routing
    router_kernel<<<TOKENS / 256, 256, 0, stream>>>(RL, counts, tlist, eidx, epos, topw);
    // 3. padded prefix offsets
    scan_kernel<<<1, 32, 0, stream>>>(counts, offsets);
    // 4. gate/up GEMM + SiLU*up fusion (worst-case grid; tiles early-exit on counts)
    gemm1_kernel<<<dim3(INTER / 128, TOKENS / 16, NUM_EXPERTS), 256, ldsBytes, stream>>>(
        X, W1, W3, counts, offsets, tlist, act);
    // 5. down-projection GEMM
    gemm2_kernel<<<dim3(HIDDEN / 128, TOKENS / 16, NUM_EXPERTS), 256, 0, stream>>>(
        act, W2, counts, offsets, y);
    // 6. deterministic weighted combine
    combine_kernel<<<dim3(HIDDEN / 4 / 256, TOKENS), 256, 0, stream>>>(
        y, eidx, epos, topw, offsets, out);
}
